// SIGN_58591943852448
// MI455X (gfx1250) — matrix operand order repaired and checked by reading, not compiled, not run on hardware
//
#include <hip/hip_runtime.h>

typedef __attribute__((ext_vector_type(16))) __bf16          v16bf;
typedef __attribute__((ext_vector_type(8)))  float           v8f;
typedef __attribute__((ext_vector_type(16))) unsigned short  v16u;
typedef __attribute__((ext_vector_type(8)))  unsigned short  v8u;

#define N_NODES 50000
#define NFEAT   256
#define NHID    64
#define HOPS    4
#define NOUT    256      // HOPS * NHID
#define NEDGES  800000

__device__ __forceinline__ unsigned short f2bf(float f) {
    unsigned int u = __builtin_bit_cast(unsigned int, f);
    u += 0x7FFFu + ((u >> 16) & 1u);          // round-to-nearest-even
    return (unsigned short)(u >> 16);
}

// ---------------- zero the output accumulator ----------------
__global__ __launch_bounds__(256) void zero_f4(float4* __restrict__ p, int n4) {
    int i = blockIdx.x * 256 + threadIdx.x;
    if (i < n4) p[i] = make_float4(0.f, 0.f, 0.f, 0.f);
}

// ---------------- fp32 -> bf16 conversion of x ----------------
__global__ __launch_bounds__(256) void conv_x(const float* __restrict__ x,
                                              unsigned short* __restrict__ xb, int n) {
    int i = blockIdx.x * 256 + threadIdx.x;
    if (i < n) xb[i] = f2bf(x[i]);
}

// ---- build W^T bf16: wt[ncol][k], ncol = hop*64 + j, from W[hop][k][j] ----
__global__ __launch_bounds__(256) void conv_w(const float* __restrict__ W,
                                              unsigned short* __restrict__ wt) {
    int ncol = blockIdx.x;            // 0..255
    int k    = threadIdx.x;           // 0..255
    int hop  = ncol >> 6, j = ncol & 63;
    wt[(size_t)ncol * NFEAT + k] =
        f2bf(W[((size_t)hop * NFEAT + k) * NHID + j]);
}

// ---------------- fused 4-hop GEMM: h = x @ Wcat + bcat -------------------
// Block = 128 threads (4 waves). Block owns one 16-row M tile; each wave owns
// 4 consecutive 16x16 N tiles (64 output columns) and reuses its A fragment
// across them: per K-step 2 A b128-loads feed 4 WMMAs.
// A fragment: row M = lane%16, K base = (lane>=16)*16,
//   low half = K base..base+7, high half = K base+8..base+15.
// B fragment: col N = lane%16, K = (lane>=16)*16 .. +15 contiguous (wt is B^T).
__global__ __launch_bounds__(128) void sign_gemm_wmma(
    const unsigned short* __restrict__ xb,   // [N_NODES][256] bf16
    const unsigned short* __restrict__ wt,   // [256][256]     bf16 (B^T)
    const float*          __restrict__ bias, // [256] = b[4][64] flat
    float*                __restrict__ h)    // [N_NODES][256] f32
{
    const int tile_m = blockIdx.x;           // 0..3124 (50000/16 exactly)
    const int wv     = threadIdx.x >> 5;     // 0..3: which group of 4 N tiles
    const int lane   = threadIdx.x & 31;

    const int m_row  = tile_m * 16 + (lane & 15);
    const int ncol0  = wv * 64 + (lane & 15);   // sub-tile nt adds nt*16
    const int kgA    = (lane >> 4) * 16;        // 0 or 16
    const int kgB    = (lane >> 4) * 16;        // 0 or 16

    const unsigned short* __restrict__ arow = xb + (size_t)m_row * NFEAT;

    // bias is uniform down a column: init every accumulator row with it
    v8f c[4];
    #pragma unroll
    for (int nt = 0; nt < 4; ++nt) {
        float bv = bias[ncol0 + nt * 16];
        #pragma unroll
        for (int v = 0; v < 8; ++v) c[nt][v] = bv;
    }

    #pragma unroll
    for (int k = 0; k < NFEAT; k += 32) {
        v8u alo = *(const v8u*)(arow + k + kgA);        // K = k+kgA+0..7
        v8u ahi = *(const v8u*)(arow + k + kgA + 8);    // K = k+kgA+8..15
        v16u au;
        #pragma unroll
        for (int j = 0; j < 8; ++j) { au[j] = alo[j]; au[j + 8] = ahi[j]; }
        v16bf a = __builtin_bit_cast(v16bf, au);

        #pragma unroll
        for (int nt = 0; nt < 4; ++nt) {
            const unsigned short* __restrict__ bcol =
                wt + (size_t)(ncol0 + nt * 16) * NFEAT;
            v16u bu = *(const v16u*)(bcol + k + kgB);   // K = k+kgB+0..15
            v16bf b = __builtin_bit_cast(v16bf, bu);
            c[nt] = __builtin_amdgcn_wmma_f32_16x16x32_bf16(
                        /*neg_a=*/false, a, /*neg_b=*/false, b,
                        /*c_mod=*/(short)0, c[nt],
                        /*reuse_a=*/false, /*reuse_b=*/false);
        }
    }

    // C layout: VGPR v -> row (lane<16 ? v : v+8), col = lane%16
    const int rbase = tile_m * 16 + ((lane >> 4) * 8);
    #pragma unroll
    for (int nt = 0; nt < 4; ++nt) {
        #pragma unroll
        for (int v = 0; v < 8; ++v)
            h[(size_t)(rbase + v) * NOUT + ncol0 + nt * 16] = c[nt][v];
    }
}

// ---------------- SpMM scatter-add: one wave per (hop, edge) --------------
__global__ __launch_bounds__(256) void sign_spmm(
    const int*   __restrict__ rows,   // [4][NEDGES]
    const int*   __restrict__ cols,   // [4][NEDGES]
    const float* __restrict__ vals,   // [4][NEDGES]
    const float* __restrict__ h,      // [N_NODES][256]
    float*       __restrict__ out)    // [N_NODES][256] accumulator
{
    const int e = blockIdx.x * 8 + (threadIdx.x >> 5);
    if (e >= NEDGES) return;
    const int hop  = blockIdx.y;
    const int lane = threadIdx.x & 31;
    const size_t eo = (size_t)hop * NEDGES + e;

    const int   r    = rows[eo];
    const int   cidx = cols[eo];
    const float v    = vals[eo];

    const float2 hv = *(const float2*)(h + (size_t)cidx * NOUT + hop * NHID + lane * 2);
    float* op = out + (size_t)r * NOUT + hop * NHID + lane * 2;
    unsafeAtomicAdd(op + 0, v * hv.x);
    unsafeAtomicAdd(op + 1, v * hv.y);
}

// ---------------- in-place ELU on the accumulator -------------------------
__global__ __launch_bounds__(256) void elu_inplace(float* __restrict__ out, int n) {
    int i = blockIdx.x * 256 + threadIdx.x;
    if (i < n) {
        float x = out[i];
        out[i] = x > 0.f ? x : (__expf(x) - 1.f);
    }
}

extern "C" void kernel_launch(void* const* d_in, const int* in_sizes, int n_in,
                              void* d_out, int out_size, void* d_ws, size_t ws_size,
                              hipStream_t stream) {
    const float* x        = (const float*)d_in[0];   // [50000][256]
    const float* W        = (const float*)d_in[1];   // [4][256][64]
    const float* b        = (const float*)d_in[2];   // [4][64]
    const int*   adj_rows = (const int*)  d_in[3];   // [4][800000]
    const int*   adj_cols = (const int*)  d_in[4];   // [4][800000]
    const float* adj_vals = (const float*)d_in[5];   // [4][800000]
    float* out = (float*)d_out;                      // [50000][256]

    // workspace layout
    char* ws = (char*)d_ws;
    unsigned short* xb = (unsigned short*)ws;                       // 25,600,000 B
    unsigned short* wt = (unsigned short*)(ws + 25600000);          //    131,072 B
    float*          h  = (float*)         (ws + 25600000 + 131072); // 51,200,000 B

    const int n_out = N_NODES * NOUT;           // 12,800,000
    const int n_x   = N_NODES * NFEAT;          // 12,800,000

    // 1) zero the atomic accumulator (d_out)
    zero_f4<<<(n_out / 4 + 255) / 256, 256, 0, stream>>>((float4*)out, n_out / 4);
    // 2) x -> bf16
    conv_x<<<(n_x + 255) / 256, 256, 0, stream>>>(x, xb, n_x);
    // 3) W -> B^T bf16
    conv_w<<<NOUT, NFEAT, 0, stream>>>(W, wt);
    // 4) fused 4-hop GEMM via WMMA: 3125 blocks x 4 waves, 32 WMMA/wave
    sign_gemm_wmma<<<N_NODES / 16, 128, 0, stream>>>(xb, wt, b, h);
    // 5) SpMM scatter-add per hop
    sign_spmm<<<dim3((NEDGES + 7) / 8, HOPS), 256, 0, stream>>>(
        adj_rows, adj_cols, adj_vals, h, out);
    // 6) ELU in place
    elu_inplace<<<(n_out + 255) / 256, 256, 0, stream>>>(out, n_out);
}
